// In2MA_1322849927556
// MI455X (gfx1250) — compile-verified, matrix operand order checked by reading, not run on hardware
//
#include <hip/hip_runtime.h>
#include <hip/hip_bf16.h>
#include <math.h>

typedef __attribute__((ext_vector_type(16))) _Float16 v16h;
typedef __attribute__((ext_vector_type(8)))  float    v8f;

#define DEVINL __device__ __forceinline__

// ---- WMMA fragment helpers (gfx1250 wave32 layouts, 16x16x32 f16) ----
// A (16x32): lane l holds row M = l%16; halves i: k = base + (i<8 ? i : i+8),
//            base = (l<16) ? 0 : 8   (K 0..7,16..23 in lanes 0-15; 8..15,24..31 in lanes 16-31)
// B (32x16): lane l holds col N = l%16; same k mapping.
// C/D f32  : lane l holds col N = l%16, rows M = 8*(l/16) + v, v=0..7.

DEVINL int frag_k(int lane, int i) {
  int base = (lane & 16) ? 8 : 0;
  return base + ((i < 8) ? i : (i + 8));
}

// A fragment from row-major MxK source
DEVINL v16h load_a_rm(const _Float16* src, int ld, int r0, int k0, int kvalid, int lane) {
  v16h a;
  int row = r0 + (lane & 15);
  #pragma unroll
  for (int i = 0; i < 16; ++i) {
    int k = frag_k(lane, i);
    a[i] = (k < kvalid) ? src[row * ld + k0 + k] : (_Float16)0.0f;
  }
  return a;
}

// A fragment from K-major source: element (row,k) at src[(k0+k)*ld + r0+row]
DEVINL v16h load_a_km(const _Float16* src, int ld, int r0, int k0, int kvalid, int lane) {
  v16h a;
  int row = r0 + (lane & 15);
  #pragma unroll
  for (int i = 0; i < 16; ++i) {
    int k = frag_k(lane, i);
    a[i] = (k < kvalid) ? src[(k0 + k) * ld + row] : (_Float16)0.0f;
  }
  return a;
}

// B fragment from row-major KxN source
DEVINL v16h load_b_kn(const _Float16* src, int ld, int k0, int n0, int kvalid, int nvalid, int lane) {
  v16h bv;
  int n = lane & 15;
  #pragma unroll
  for (int i = 0; i < 16; ++i) {
    int k = frag_k(lane, i);
    bv[i] = (k < kvalid && n < nvalid) ? src[(k0 + k) * ld + n0 + n] : (_Float16)0.0f;
  }
  return bv;
}

// B fragment from row-major NxK source (e.g. weight W[N][K], computing X @ W^T)
DEVINL v16h load_b_nk(const _Float16* src, int ld, int k0, int n0, int kvalid, int nvalid, int lane) {
  v16h bv;
  int n = lane & 15;
  #pragma unroll
  for (int i = 0; i < 16; ++i) {
    int k = frag_k(lane, i);
    bv[i] = (k < kvalid && n < nvalid) ? src[(n0 + n) * ld + k0 + k] : (_Float16)0.0f;
  }
  return bv;
}

DEVINL v8f wmma16(v16h a, v16h b, v8f c) {
  return __builtin_amdgcn_wmma_f32_16x16x32_f16(false, a, false, b, (short)0, c, false, false);
}

DEVINL void store_c_f16(_Float16* dst, int ld, int r0, int n0, v8f acc, int lane) {
  int n  = n0 + (lane & 15);
  int m0 = r0 + ((lane & 16) ? 8 : 0);
  #pragma unroll
  for (int v = 0; v < 8; ++v) dst[(m0 + v) * ld + n] = (_Float16)acc[v];
}

// generic C = A(MxK, f16 LDS) @ W^T(W is NxK f16), C f16, tiles strided across 8 waves
DEVINL void gemm_xwT(const _Float16* A, int lda, const _Float16* W, int ldw,
                     _Float16* C, int ldc, int M, int N, int K, int wave, int lane) {
  int nt = N >> 4, mt = M >> 4;
  for (int t = wave; t < mt * nt; t += 8) {
    int mi = t / nt, ni = t - mi * nt;
    v8f acc = {};
    for (int k0 = 0; k0 < K; k0 += 32) {
      v16h a  = load_a_rm(A, lda, mi << 4, k0, 32, lane);
      v16h bv = load_b_nk(W, ldw, k0, ni << 4, 32, 16, lane);
      acc = wmma16(a, bv, acc);
    }
    store_c_f16(C, ldc, mi << 4, ni << 4, acc, lane);
  }
}

DEVINL float redmax16(float x) {
  x = fmaxf(x, __shfl_xor(x, 1, 32));
  x = fmaxf(x, __shfl_xor(x, 2, 32));
  x = fmaxf(x, __shfl_xor(x, 4, 32));
  x = fmaxf(x, __shfl_xor(x, 8, 32));
  return x;
}
DEVINL float redsum16(float x) {
  x += __shfl_xor(x, 1, 32);
  x += __shfl_xor(x, 2, 32);
  x += __shfl_xor(x, 4, 32);
  x += __shfl_xor(x, 8, 32);
  return x;
}

// -------------------- weight f32 -> f16 conversion --------------------
__global__ void in2ma_cvt_f16(const float* __restrict__ src, _Float16* __restrict__ dst, int n) {
  int i = blockIdx.x * 256 + threadIdx.x;
  if (i < n) dst[i] = (_Float16)src[i];
}

// -------------------- fused per-window kernel --------------------
// grid = 4096 (B=4 * 32 * 32 windows), block = 256 (8 waves == 8 heads)
__global__ __launch_bounds__(256, 1)
void in2ma_fused(const float* __restrict__ x, const float* __restrict__ pan,
                 const _Float16* __restrict__ wts,
                 const float* __restrict__ pos_inner, const float* __restrict__ pos_color,
                 float* __restrict__ out) {
  // LDS layout (halfs), phase-aliased; total 26624 halfs = 52 KB
  __shared__ _Float16 smem[26624];
  _Float16* sXF   = smem;            // 4096  phase A input, later CAT
  _Float16* sCAT  = smem;
  _Float16* sPAN  = smem + 4096;     // 2048  phase A input, later K2
  _Float16* sK2   = smem + 4096;
  _Float16* sPQ   = smem + 6144;     // 2048  pan_q (lives to the end)
  _Float16* sPK   = smem + 8192;     // 2048  pan_k, later cos (f32)
  float*    sCos  = (float*)(smem + 8192);
  _Float16* sV1   = smem + 10240;    // 4096  x_v1, later OUT
  _Float16* sOUT  = smem + 10240;
  _Float16* sQ1C  = smem + 14336;    // 2048
  _Float16* sK1C  = smem + 16384;    // 2048  (Q1C+K1C region later OUT2)
  _Float16* sOUT2 = smem + 14336;
  _Float16* sV2   = smem + 18432;    // 4096
  _Float16* sStg  = smem + 22528;    // 4096: 512 halfs (16x32 tile) per wave

  const int tid  = threadIdx.x;
  const int lane = tid & 31;
  const int wave = tid >> 5;

  const int wid = blockIdx.x;
  const int b   = wid >> 10;
  const int rem = wid & 1023;
  const int y0  = (rem >> 5) << 3;
  const int x0  = (rem & 31) << 3;

  // ---- stage window into LDS (f16) : xf[t][c], pan[t][c] ----
  for (int idx = tid; idx < 4096; idx += 256) {
    int c = idx >> 6, t = idx & 63;
    sXF[t * 64 + c] = (_Float16)x[((b * 64 + c) * 256 + y0 + (t >> 3)) * 256 + x0 + (t & 7)];
  }
  for (int idx = tid; idx < 2048; idx += 256) {
    int c = idx >> 6, t = idx & 63;
    sPAN[t * 32 + c] = (_Float16)pan[((b * 32 + c) * 256 + y0 + (t >> 3)) * 256 + x0 + (t & 7)];
  }
  __syncthreads();

  const _Float16* W_pq = wts;            // 32x32
  const _Float16* W_pk = wts + 1024;     // 32x32
  const _Float16* W_v1 = wts + 2048;     // 64x64
  const _Float16* W_v2 = wts + 6144;     // 64x64
  const _Float16* W_k2 = wts + 10240;    // 32x64
  const _Float16* W_q1 = wts + 12288;    // 32x64
  const _Float16* W_k1 = wts + 14336;    // 32x64
  const _Float16* W_io = wts + 16384;    // 64x64
  const _Float16* W_to = wts + 20480;    // 64x64

  // ---- phase A: projections ----
  gemm_xwT(sPAN, 32, W_pq, 32, sPQ,  32, 64, 32, 32, wave, lane);
  gemm_xwT(sPAN, 32, W_pk, 32, sPK,  32, 64, 32, 32, wave, lane);
  gemm_xwT(sXF,  64, W_v1, 64, sV1,  64, 64, 64, 64, wave, lane);
  gemm_xwT(sXF,  64, W_q1, 64, sQ1C, 32, 64, 32, 64, wave, lane);
  gemm_xwT(sXF,  64, W_k1, 64, sK1C, 32, 64, 32, 64, wave, lane);
  __syncthreads();

  const int   h    = wave;               // wave == head
  const int   n16  = lane & 15;
  const int   mb   = (lane & 16) ? 8 : 0;
  const float scl  = 0.35355339059327373f;   // 8^-0.5 (both inner & color scales)
  _Float16*   st   = sStg + wave * 512;

  // ---- pan window attention (seq=64, head_dim=4, K zero-padded) ----
  {
    v16h bk[4], bv[2];
    #pragma unroll
    for (int nj = 0; nj < 4; ++nj)          // B = K^T : b[d][j] = pan_k[j][h*4+d]
      bk[nj] = load_b_nk(sPK, 32, h * 4, nj * 16, 4, 16, lane);
    #pragma unroll
    for (int kt = 0; kt < 2; ++kt)          // B = v1_pan head slice (64x4)
      bv[kt] = load_b_kn(sV1, 64, kt * 32, h * 4, 32, 4, lane);

    for (int mi = 0; mi < 4; ++mi) {
      v16h aq = load_a_rm(sPQ, 32, mi * 16, h * 4, 4, lane);
      v8f  S[4];
      #pragma unroll
      for (int nj = 0; nj < 4; ++nj) { v8f z = {}; S[nj] = wmma16(aq, bk[nj], z); }

      #pragma unroll
      for (int v = 0; v < 8; ++v) {         // row-wise softmax (+pos bias)
        int   i = mi * 16 + mb + v;
        float m = -3.0e38f;
        #pragma unroll
        for (int nj = 0; nj < 4; ++nj) {
          float s = S[nj][v] * scl + pos_inner[(h * 64 + i) * 64 + nj * 16 + n16];
          S[nj][v] = s;
          m = fmaxf(m, s);
        }
        m = redmax16(m);
        float sum = 0.f;
        #pragma unroll
        for (int nj = 0; nj < 4; ++nj) { float e = __expf(S[nj][v] - m); S[nj][v] = e; sum += e; }
        sum = redsum16(sum);
        float inv = 1.0f / sum;
        #pragma unroll
        for (int nj = 0; nj < 4; ++nj) S[nj][v] *= inv;
      }

      v8f O = {};
      #pragma unroll
      for (int kt = 0; kt < 2; ++kt) {      // P @ V via restage of P as A-fragment
        #pragma unroll
        for (int hf = 0; hf < 2; ++hf) {
          int nj = kt * 2 + hf;
          #pragma unroll
          for (int v = 0; v < 8; ++v) st[(mb + v) * 32 + hf * 16 + n16] = (_Float16)S[nj][v];
        }
        asm volatile("s_wait_dscnt 0" ::: "memory");
        v16h pa = load_a_rm(st, 32, 0, 0, 32, lane);
        O = wmma16(pa, bv[kt], O);
      }
      if (n16 < 4) {
        #pragma unroll
        for (int v = 0; v < 8; ++v)
          sCAT[(mi * 16 + mb + v) * 64 + h * 4 + n16] = (_Float16)O[v];
      }
    }
  }

  // ---- color attention (32 positions, head_dim=8) ----
  {
    v16h bk[2];
    #pragma unroll
    for (int nj = 0; nj < 2; ++nj)          // b[e][p'] = k1c[h*8+e][p']
      bk[nj] = load_b_kn(sK1C, 32, h * 8, nj * 16, 8, 16, lane);
    // b[p'][e] = v1_color[h*8+e][p']  (v1_color = cols 32.. of x_v1)
    v16h bvc = load_b_nk(sV1 + 32, 64, 0, h * 8, 32, 8, lane);

    for (int mi = 0; mi < 2; ++mi) {
      v16h aq = load_a_km(sQ1C, 32, mi * 16, h * 8, 8, lane);  // a[p][e] = q1c[h*8+e][p]
      v8f  S[2];
      #pragma unroll
      for (int nj = 0; nj < 2; ++nj) { v8f z = {}; S[nj] = wmma16(aq, bk[nj], z); }

      #pragma unroll
      for (int v = 0; v < 8; ++v) {
        int   i = mi * 16 + mb + v;
        float m = -3.0e38f;
        #pragma unroll
        for (int nj = 0; nj < 2; ++nj) {
          float s = S[nj][v] * scl + pos_color[(h * 32 + i) * 32 + nj * 16 + n16];
          S[nj][v] = s;
          m = fmaxf(m, s);
        }
        m = redmax16(m);
        float sum = 0.f;
        #pragma unroll
        for (int nj = 0; nj < 2; ++nj) { float e = __expf(S[nj][v] - m); S[nj][v] = e; sum += e; }
        sum = redsum16(sum);
        float inv = 1.0f / sum;
        #pragma unroll
        for (int nj = 0; nj < 2; ++nj) S[nj][v] *= inv;
      }

      #pragma unroll
      for (int hf = 0; hf < 2; ++hf) {
        #pragma unroll
        for (int v = 0; v < 8; ++v) st[(mb + v) * 32 + hf * 16 + n16] = (_Float16)S[hf][v];
      }
      asm volatile("s_wait_dscnt 0" ::: "memory");
      v16h pa = load_a_rm(st, 32, 0, 0, 32, lane);
      v8f  z = {};
      v8f  O = wmma16(pa, bvc, z);
      if (n16 < 8) {                         // out_color[h*8+e][p] = O[p][e]
        #pragma unroll
        for (int v = 0; v < 8; ++v)
          sCAT[(h * 8 + n16) * 64 + 32 + mi * 16 + mb + v] = (_Float16)O[v];
      }
    }
  }
  __syncthreads();

  // ---- inner out + v2/k2 projections ----
  gemm_xwT(sCAT, 64, W_io, 64, sOUT, 64, 64, 64, 64, wave, lane);
  __syncthreads();
  gemm_xwT(sOUT, 64, W_v2, 64, sV2, 64, 64, 64, 64, wave, lane);
  gemm_xwT(sOUT, 64, W_k2, 64, sK2, 32, 64, 32, 64, wave, lane);
  __syncthreads();

  // ---- inter attention: cosine gate per (token, head) ----
  for (int idx = tid; idx < 512; idx += 256) {
    int   i = idx & 63, hh = idx >> 6;
    float dot = 0.f, nq = 0.f, nk = 0.f;
    #pragma unroll
    for (int d = 0; d < 4; ++d) {
      float q  = (float)sPQ[i * 32 + hh * 4 + d];
      float kk = (float)sK2[i * 32 + hh * 4 + d];
      dot += q * kk; nq += q * q; nk += kk * kk;
    }
    sCos[hh * 64 + i] = dot / sqrtf(nq * nk);
  }
  __syncthreads();
  for (int idx = tid; idx < 4096; idx += 256) {
    int i = idx >> 6, c = idx & 63;
    sOUT2[i * 64 + c] = (_Float16)((float)sV2[i * 64 + c] * sCos[(c >> 3) * 64 + i]);
  }
  __syncthreads();

  // ---- final projection, scatter to (b, cc, h, w) ----
  for (int t = wave; t < 16; t += 8) {
    int mi = t >> 2, ni = t & 3;
    v8f acc = {};
    #pragma unroll
    for (int k0 = 0; k0 < 64; k0 += 32) {
      v16h a  = load_a_rm(sOUT2, 64, mi * 16, k0, 32, lane);
      v16h bw = load_b_nk(W_to, 64, k0, ni * 16, 32, 16, lane);
      acc = wmma16(a, bw, acc);
    }
    int cc = ni * 16 + n16;
    #pragma unroll
    for (int v = 0; v < 8; ++v) {
      int tok = mi * 16 + mb + v;
      out[((b * 64 + cc) * 256 + y0 + (tok >> 3)) * 256 + x0 + (tok & 7)] = acc[v];
    }
  }
}

extern "C" void kernel_launch(void* const* d_in, const int* in_sizes, int n_in,
                              void* d_out, int out_size, void* d_ws, size_t ws_size,
                              hipStream_t stream) {
  const float* x   = (const float*)d_in[0];
  const float* pan = (const float*)d_in[1];
  const float* pos_inner = (const float*)d_in[11];
  const float* pos_color = (const float*)d_in[12];
  _Float16* wts = (_Float16*)d_ws;   // 24576 halfs = 48 KB of scratch

  struct Cv { int src; int off; int n; };
  const Cv cv[9] = {
    {2,     0, 1024},   // W_pan_q   32x32
    {3,  1024, 1024},   // W_pan_k   32x32
    {4,  2048, 4096},   // W_v1      64x64
    {5,  6144, 4096},   // W_v2      64x64
    {6, 10240, 2048},   // W_k2      32x64
    {7, 12288, 2048},   // W_q1c     32x64
    {8, 14336, 2048},   // W_k1c     32x64
    {9, 16384, 4096},   // W_inner_out 64x64
    {10,20480, 4096},   // W_inter_out 64x64
  };
  for (int i = 0; i < 9; ++i)
    in2ma_cvt_f16<<<(cv[i].n + 255) / 256, 256, 0, stream>>>(
        (const float*)d_in[cv[i].src], wts + cv[i].off, cv[i].n);

  in2ma_fused<<<4096, 256, 0, stream>>>(x, pan, wts, pos_inner, pos_color, (float*)d_out);
}
